// Transducer_64845416235401
// MI455X (gfx1250) — matrix operand order repaired and checked by reading, not compiled, not run on hardware
//
#include <hip/hip_runtime.h>
#include <hip/hip_bf16.h>

// Problem constants (from reference): D=512, V=8192, B=2, T=256, U=64
#define D_ 512
#define V_ 8192
#define B_ 2
#define T_ 256
#define U_ 64
#define M_TOTAL (B_ * T_ * U_)   // 32768 joint rows

typedef __attribute__((ext_vector_type(16))) __bf16       v16bf;
typedef __attribute__((ext_vector_type(8)))  float        v8f;
typedef __attribute__((ext_vector_type(4)))  unsigned int v4u;

// Trivial union (ext_vector types are POD) to assemble WMMA operands from two
// 16-byte loads.
union Frag {
    v4u   u[2];
    v16bf v;
};

// ---------------------------------------------------------------------------
// Small pointwise linears: Y(rows,512) = X(rows,512) @ W(512,512) + bias.
// 134 MFLOP worst case -> plain VALU, coalesced on W columns across lanes.
// ---------------------------------------------------------------------------
__global__ void __launch_bounds__(256)
linear512(const float* __restrict__ X, const float* __restrict__ W,
          const float* __restrict__ bias, float* __restrict__ Y, int rows) {
    int idx = blockIdx.x * 256 + threadIdx.x;
    if (idx >= rows * D_) return;
    int r = idx >> 9;
    int c = idx & (D_ - 1);
    const float* x = X + (size_t)r * D_;
    float acc = bias[c];
#pragma unroll 8
    for (int k = 0; k < D_; ++k)
        acc = fmaf(x[k], W[(size_t)k * D_ + c], acc);
    Y[idx] = acc;
}

// ---------------------------------------------------------------------------
// Repack Wv (D,V) f32 -> bf16 in WMMA B-operand order.
// Packed layout: [(nt*16 + kt)*512 + lane*16 + e], where for lane l,
// element e:  k = kt*32 + (l>>4)*16 + e,  n = nt*16 + (l&15).
// (this code's 16-bit B 32x16 order: lanes 0-15 hold K=0..15, lanes 16-31 hold K=16..31,
//  column = lane&15 ; A uses the same k order.)
// Each GEMM lane then reads its 16 bf16 as 32 contiguous bytes.
// ---------------------------------------------------------------------------
__global__ void __launch_bounds__(256)
repack_wv(const float* __restrict__ Wv, __hip_bfloat16* __restrict__ out) {
    int idx  = blockIdx.x * 256 + threadIdx.x;   // 0 .. 512*8192-1
    int e    = idx & 15;
    int lane = (idx >> 4) & 31;
    int kt   = (idx >> 9) & 15;
    int nt   = idx >> 13;
    int k = kt * 32 + (lane >> 4) * 16 + e;
    int n = nt * 16 + (lane & 15);
    out[idx] = __float2bfloat16(Wv[(size_t)k * V_ + n]);
}

// ---------------------------------------------------------------------------
// Joint GEMM: logits[m, n] = relu(enc[b,t,:] + dec[b,u,:]) @ Wv + bv
//
// Block = 256 threads (8 waves). M-tile = 32 rows (two 16-row WMMA tiles
// sharing every B fragment), N-block = 512 (one 32x64 strip per wave).
// Per K-step each wave runs 8 WMMAs against 8 global b128 (B) + 4 ds b128 (A).
// A and B fragments are double-buffered so the kt+1 loads stay outstanding
// while the kt WMMAs issue (breaks the load -> s_wait_loadcnt 0 -> wmma
// serialization seen in the previous round's asm).
//
// h tile staged in LDS as bf16 with row stride 520 elements (1040 B =
// 260 dwords; 260 % 64 = 4 -> the 16 rows touched by one b128 gather land on
// distinct banks).
// ---------------------------------------------------------------------------
#define LDS_STRIDE 520
#define MT 32

__global__ void __launch_bounds__(256)
joint_gemm(const float* __restrict__ enc, const float* __restrict__ dec,
           const __hip_bfloat16* __restrict__ wvp, const float* __restrict__ bv,
           float* __restrict__ out) {
    __shared__ __hip_bfloat16 hl[MT * LDS_STRIDE];

    const int tid = threadIdx.x;
    const int m0  = blockIdx.x * MT;    // joint-row tile base
    const int n0  = blockIdx.y * 512;   // vocab block base

    // ---- Phase A: build bf16 h tile (32 rows x 512) in LDS ----
#pragma unroll
    for (int it = 0; it < 64; ++it) {
        int i   = it * 256 + tid;       // 0..16383
        int row = i >> 9;
        int k   = i & (D_ - 1);
        int m   = m0 + row;
        int b   = m >> 14;              // / (T*U)
        int t   = (m >> 6) & (T_ - 1);
        int u   = m & (U_ - 1);
        float e = enc[((size_t)(b * T_ + t)) * D_ + k];
        float d = dec[((size_t)(b * U_ + u)) * D_ + k];
        float h = e + d;
        h = h > 0.0f ? h : 0.0f;        // relu
        hl[row * LDS_STRIDE + k] = __float2bfloat16(h);
    }
    __syncthreads();

    // ---- Phase B: WMMA over K = 512 in 16 steps of 32, double-buffered ----
    const int lane = tid & 31;
    const int wave = tid >> 5;
    const int nw   = n0 + wave * 64;    // this wave's 32x64 strip

    v8f acc[8] = {v8f{}, v8f{}, v8f{}, v8f{}, v8f{}, v8f{}, v8f{}, v8f{}};

    const int arow = lane & 15;         // A: row = lane&15
    const int ah   = lane >> 4;         // A: K-half select
    const unsigned short* hls = (const unsigned short*)hl;
    const int aoff0 = arow * LDS_STRIDE + ah * 16;              // M tile 0
    const int aoff1 = (arow + 16) * LDS_STRIDE + ah * 16;       // M tile 1

    // B base for this wave's 4 N-tiles; tile (nt,kt) = 1 KB contiguous,
    // lane reads 32 contiguous bytes; nt stride 8192 elems, kt stride 512.
    const unsigned short* bbase =
        (const unsigned short*)wvp + (size_t)(nw >> 4) * 8192 + lane * 16;

    Frag bcur[4], bnxt[4], acur0, acur1, anxt0, anxt1;

    // Preload kt = 0 operands.
#pragma unroll
    for (int j = 0; j < 4; ++j) {
        const unsigned short* bp = bbase + j * 8192;
        bcur[j].u[0] = *(const v4u*)(bp);
        bcur[j].u[1] = *(const v4u*)(bp + 8);
    }
    acur0.u[0] = *(const v4u*)(hls + aoff0);
    acur0.u[1] = *(const v4u*)(hls + aoff0 + 8);
    acur1.u[0] = *(const v4u*)(hls + aoff1);
    acur1.u[1] = *(const v4u*)(hls + aoff1 + 8);

#pragma unroll
    for (int kt = 0; kt < 16; ++kt) {
        // Issue next-iteration loads first so they stay outstanding across
        // the 8 WMMAs below (last iteration wraps to kt=0: harmless).
        const int ktn = (kt + 1) & 15;
#pragma unroll
        for (int j = 0; j < 4; ++j) {
            const unsigned short* bp = bbase + (size_t)ktn * 512 + j * 8192;
            bnxt[j].u[0] = *(const v4u*)(bp);
            bnxt[j].u[1] = *(const v4u*)(bp + 8);
        }
        anxt0.u[0] = *(const v4u*)(hls + aoff0 + ktn * 32);
        anxt0.u[1] = *(const v4u*)(hls + aoff0 + ktn * 32 + 8);
        anxt1.u[0] = *(const v4u*)(hls + aoff1 + ktn * 32);
        anxt1.u[1] = *(const v4u*)(hls + aoff1 + ktn * 32 + 8);

#pragma unroll
        for (int j = 0; j < 4; ++j) {
            acc[j] = __builtin_amdgcn_wmma_f32_16x16x32_bf16(
                false, acur0.v, false, bcur[j].v, (short)0, acc[j], false, false);
            acc[4 + j] = __builtin_amdgcn_wmma_f32_16x16x32_bf16(
                false, acur1.v, false, bcur[j].v, (short)0, acc[4 + j], false, false);
        }

#pragma unroll
        for (int j = 0; j < 4; ++j) bcur[j] = bnxt[j];
        acur0 = anxt0;
        acur1 = anxt1;
    }

    // ---- Epilogue: + bias, store f32 logits ----
    const int ncol = lane & 15;
    const int mh   = lane >> 4;         // C: lanes 16-31 hold M = 8+r
#pragma unroll
    for (int j = 0; j < 4; ++j) {
        int n    = nw + j * 16 + ncol;
        float bb = bv[n];
#pragma unroll
        for (int mt = 0; mt < 2; ++mt) {
#pragma unroll
            for (int r = 0; r < 8; ++r) {
                int m = m0 + mt * 16 + mh * 8 + r;
                out[(size_t)m * V_ + n] = acc[mt * 4 + j][r] + bb;
            }
        }
    }
}

// ---------------------------------------------------------------------------
// In-place log-softmax over rows of 8192. One block per row; 32 values per
// thread held in registers (8 x float4), two LDS tree reductions.
// ---------------------------------------------------------------------------
__global__ void __launch_bounds__(256)
logsoftmax_inplace(float* __restrict__ out) {
    const size_t row = blockIdx.x;
    float4* p = (float4*)(out + row * (size_t)V_);
    const int tid = threadIdx.x;

    float4 vals[8];
    float mx = -3.402823466e38f;
#pragma unroll
    for (int i = 0; i < 8; ++i) {
        vals[i] = p[tid + i * 256];
        mx = fmaxf(mx, fmaxf(fmaxf(vals[i].x, vals[i].y),
                             fmaxf(vals[i].z, vals[i].w)));
    }

    __shared__ float red[256];
    red[tid] = mx;
    __syncthreads();
    for (int s = 128; s > 0; s >>= 1) {
        if (tid < s) red[tid] = fmaxf(red[tid], red[tid + s]);
        __syncthreads();
    }
    mx = red[0];
    __syncthreads();

    float sum = 0.0f;
#pragma unroll
    for (int i = 0; i < 8; ++i) {
        sum += __expf(vals[i].x - mx) + __expf(vals[i].y - mx) +
               __expf(vals[i].z - mx) + __expf(vals[i].w - mx);
    }
    red[tid] = sum;
    __syncthreads();
    for (int s = 128; s > 0; s >>= 1) {
        if (tid < s) red[tid] += red[tid + s];
        __syncthreads();
    }
    const float lse = mx + __logf(red[0]);

#pragma unroll
    for (int i = 0; i < 8; ++i) {
        float4 v = vals[i];
        v.x -= lse; v.y -= lse; v.z -= lse; v.w -= lse;
        p[tid + i * 256] = v;
    }
}

// ---------------------------------------------------------------------------
// Launcher. Workspace layout (9.25 MB total):
//   enc  f32 512x512  @ 0          (1 MiB)
//   dec  f32 128x512  @ 1 MiB      (256 KiB)
//   WvPk bf16 512x8192 @ 1.25 MiB  (8 MiB)
// ---------------------------------------------------------------------------
extern "C" void kernel_launch(void* const* d_in, const int* in_sizes, int n_in,
                              void* d_out, int out_size, void* d_ws, size_t ws_size,
                              hipStream_t stream) {
    const float* v  = (const float*)d_in[0];
    const float* t  = (const float*)d_in[1];
    const float* W1 = (const float*)d_in[2];
    const float* b1 = (const float*)d_in[3];
    const float* W2 = (const float*)d_in[4];
    const float* b2 = (const float*)d_in[5];
    const float* Wv = (const float*)d_in[6];
    const float* bv = (const float*)d_in[7];
    float* out = (float*)d_out;

    char* ws = (char*)d_ws;
    float*          enc = (float*)ws;                                  // 512*512
    float*          dec = (float*)(ws + (size_t)512 * 512 * 4);        // 128*512
    __hip_bfloat16* wvp = (__hip_bfloat16*)(ws + (size_t)512 * 512 * 4
                                               + (size_t)128 * 512 * 4);

    linear512<<<(512 * 512) / 256, 256, 0, stream>>>(v, W1, b1, enc, 512);
    linear512<<<(128 * 512) / 256, 256, 0, stream>>>(t, W2, b2, dec, 128);
    repack_wv<<<(512 * 8192) / 256, 256, 0, stream>>>(Wv, wvp);

    dim3 grid(M_TOTAL / MT, V_ / 512);   // 1024 x 16 blocks
    joint_gemm<<<grid, 256, 0, stream>>>(enc, dec, wvp, bv, out);

    logsoftmax_inplace<<<M_TOTAL, 256, 0, stream>>>(out);
}
